// VQ_StraightThrough_8074538516849
// MI455X (gfx1250) — matrix operand order repaired and checked by reading, not compiled, not run on hardware
//
#include <hip/hip_runtime.h>
#include <hip/hip_bf16.h>

typedef __bf16 v16bf __attribute__((ext_vector_type(16)));
typedef __bf16 v8bf  __attribute__((ext_vector_type(8)));
typedef float  v8f   __attribute__((ext_vector_type(8)));
typedef float  v4f   __attribute__((ext_vector_type(4)));

#define NB    16
#define CDIM  256
#define HDIM  64
#define WDIM  64
#define NCODE 1024
#define HW    (HDIM * WDIM)
#define LDSROW 264   // 256 + 8 bf16 pad to spread LDS banks

__global__ void vq_zero_loss(float* __restrict__ loss) {
    if (threadIdx.x < NB) loss[threadIdx.x] = 0.0f;
}

// Convert codebook fp32 -> bf16 (workspace) and compute ||e_k||^2 in fp32.
__global__ __launch_bounds__(CDIM) void vq_prep(const float* __restrict__ ew,
                                                __bf16* __restrict__ ewbf,
                                                float* __restrict__ enorm) {
    const int k = blockIdx.x;
    const int c = threadIdx.x;
    const float v = ew[(size_t)k * CDIM + c];
    ewbf[(size_t)k * CDIM + c] = (__bf16)v;
    __shared__ float red[CDIM];
    red[c] = v * v;
    __syncthreads();
    for (int s = CDIM / 2; s > 0; s >>= 1) {
        if (c < s) red[c] += red[c + s];
        __syncthreads();
    }
    if (c == 0) enorm[k] = red[0];
}

// Load eight 32x16 bf16 B tiles (K=256) (contiguous k order, matching A: lane n holds code n's 16
// contiguous K values; half-waves split K 0..15 / 16..31 per 32-chunk).
__device__ __forceinline__ void load_btile(v16bf (&bt)[8], const __bf16* base) {
#pragma unroll
    for (int kk = 0; kk < 8; ++kk) {
        union { v16bf v; v8bf h2[2]; } u;
        u.h2[0] = *(const v8bf*)(base + kk * 32);
        u.h2[1] = *(const v8bf*)(base + kk * 32 + 8);
        bt[kk] = u.v;
    }
}

// 8-chunk WMMA chain (K=256) + running argmin update for one 16-code tile.
__device__ __forceinline__ void compute_step(const v16bf (&a)[8], const v16bf (&bt)[8],
                                             int code, const float* __restrict__ enorm,
                                             float (&best)[8], int (&bidx)[8]) {
    const float en = enorm[code];       // issue early; hides under WMMA chain
    v8f acc = {};
#pragma unroll
    for (int kk = 0; kk < 8; ++kk) {
        acc = __builtin_amdgcn_wmma_f32_16x16x32_bf16(
            false, a[kk], false, bt[kk], (short)0, acc, false, false);
    }
#pragma unroll
    for (int r = 0; r < 8; ++r) {
        const float d = en - 2.0f * acc[r];
        if (d < best[r] || (d == best[r] && code < bidx[r])) {
            best[r] = d; bidx[r] = code;
        }
    }
}

__global__ __launch_bounds__(256) void vq_main(const float* __restrict__ ze,
                                               const float* __restrict__ ew,
                                               const __bf16* __restrict__ ewbf,
                                               const float* __restrict__ enorm,
                                               float* __restrict__ out,
                                               float* __restrict__ loss) {
    __shared__ __bf16 zbuf[WDIM * LDSROW];   // 64 positions x 256 features (bf16)
    __shared__ float  redD[WDIM * 2];
    __shared__ int    redI[WDIM * 2];
    __shared__ int    widx[WDIM];
    __shared__ float  lred[256];

    const int t  = threadIdx.x;
    const int bh = blockIdx.x;
    const int b  = bh >> 6;
    const int h  = bh & 63;
    const float* zb = ze + (size_t)b * CDIM * HW + (size_t)h * WDIM;

    // ---- Phase 1: stage z_e (b,h) row into LDS as bf16, row-major [w][c] ----
    {
        const int w4 = (t & 15) * 4;
        const int cq = t >> 4;               // 0..15
        for (int c0 = 0; c0 < CDIM; c0 += 16) {
            const int c = c0 + cq;
            const v4f z4 = *(const v4f*)(zb + (size_t)c * HW + w4);
            zbuf[(w4 + 0) * LDSROW + c] = (__bf16)z4[0];
            zbuf[(w4 + 1) * LDSROW + c] = (__bf16)z4[1];
            zbuf[(w4 + 2) * LDSROW + c] = (__bf16)z4[2];
            zbuf[(w4 + 3) * LDSROW + c] = (__bf16)z4[3];
        }
    }
    __syncthreads();

    // ---- Phase 2: WMMA distances + running argmin (double-buffered B) ----
    {
        const int lane   = t & 31;
        const int wid    = t >> 5;
        const int mbase  = (wid & 3) * 16;    // 4 M-subtiles of 16 positions
        const int nstart = (wid >> 2) * 512;  // 2 N-halves of 512 codes
        const int hi     = lane >> 4;         // half-wave select
        const int col    = lane & 15;

        // A tile: 16 rows x K=256, contiguous k = 16*hi + s per 32-chunk, resident in VGPRs.
        v16bf a[8];
        {
            const __bf16* arow = &zbuf[(mbase + col) * LDSROW];
#pragma unroll
            for (int kk = 0; kk < 8; ++kk) {
                const int k0 = kk * 32 + hi * 16;
                union { v16bf v; v8bf h2[2]; } u;
                u.h2[0] = *(const v8bf*)(arow + k0);
                u.h2[1] = *(const v8bf*)(arow + k0 + 8);
                a[kk] = u.v;
            }
        }

        float best[8];
        int   bidx[8];
#pragma unroll
        for (int r = 0; r < 8; ++r) { best[r] = 3.4e38f; bidx[r] = 0x7fffffff; }

        // Per-lane base pointer into bf16 codebook for this wave's N-half.
        const __bf16* bbase = ewbf + (size_t)(nstart + col) * CDIM + hi * 16;
        const size_t  nstr  = (size_t)16 * CDIM;   // 16 codes per chunk

        v16bf b0[8], b1[8];
        load_btile(b0, bbase);
        for (int nc = 0; nc < 32; nc += 2) {
            // prefetch nc+1 while computing nc
            load_btile(b1, bbase + (size_t)(nc + 1) * nstr);
            compute_step(a, b0, nstart + nc * 16 + col, enorm, best, bidx);
            // prefetch nc+2 while computing nc+1 (clamped; last reload is dead)
            const int nn = (nc + 2 < 32) ? (nc + 2) : nc;
            load_btile(b0, bbase + (size_t)nn * nstr);
            compute_step(a, b1, nstart + (nc + 1) * 16 + col, enorm, best, bidx);
        }

        // Reduce over 16 column slots within each half-wave.
        for (int m = 1; m <= 8; m <<= 1) {
#pragma unroll
            for (int r = 0; r < 8; ++r) {
                const float od = __shfl_xor(best[r], m, 32);
                const int   oi = __shfl_xor(bidx[r], m, 32);
                if (od < best[r] || (od == best[r] && oi < bidx[r])) {
                    best[r] = od; bidx[r] = oi;
                }
            }
        }
        if (col == 0) {
            const int wn = wid >> 2;
#pragma unroll
            for (int r = 0; r < 8; ++r) {
                const int row = mbase + hi * 8 + r;   // D layout: VGPR r, halves
                redD[row * 2 + wn] = best[r];
                redI[row * 2 + wn] = bidx[r];
            }
        }
    }
    __syncthreads();

    if (t < WDIM) {
        const float d0 = redD[t * 2], d1 = redD[t * 2 + 1];
        const int   i0 = redI[t * 2], i1 = redI[t * 2 + 1];
        widx[t] = (d1 < d0 || (d1 == d0 && i1 < i0)) ? i1 : i0;
    }
    __syncthreads();

    // ---- Phase 3: gather z_q, write [B,C,H,W] output, accumulate loss ----
    float lacc = 0.0f;
    {
        const int w4 = (t & 15) * 4;
        const int cq = t >> 4;
        const float* e0 = ew + (size_t)widx[w4 + 0] * CDIM;
        const float* e1 = ew + (size_t)widx[w4 + 1] * CDIM;
        const float* e2 = ew + (size_t)widx[w4 + 2] * CDIM;
        const float* e3 = ew + (size_t)widx[w4 + 3] * CDIM;
        float* ob = out + (size_t)b * CDIM * HW + (size_t)h * WDIM;
        for (int c0 = 0; c0 < CDIM; c0 += 16) {
            const int c = c0 + cq;
            const v4f z4 = *(const v4f*)(zb + (size_t)c * HW + w4);
            v4f e4;
            e4[0] = e0[c]; e4[1] = e1[c]; e4[2] = e2[c]; e4[3] = e3[c];
            *(v4f*)(ob + (size_t)c * HW + w4) = e4;     // fwd value == z_q
            const v4f df = e4 - z4;
            lacc += df[0] * df[0] + df[1] * df[1] + df[2] * df[2] + df[3] * df[3];
        }
    }
    lred[t] = lacc;
    __syncthreads();
    for (int s = 128; s > 0; s >>= 1) {
        if (t < s) lred[t] += lred[t + s];
        __syncthreads();
    }
    if (t == 0) {
        // vq_loss[b] = (1 + 0.25) * mean over (H,W,C)
        atomicAdd(&loss[b], lred[0] * (1.25f / (float)(HW * CDIM)));
    }
}

extern "C" void kernel_launch(void* const* d_in, const int* in_sizes, int n_in,
                              void* d_out, int out_size, void* d_ws, size_t ws_size,
                              hipStream_t stream) {
    const float* ze = (const float*)d_in[0];   // [16, 256, 64, 64] fp32
    const float* ew = (const float*)d_in[1];   // [1024, 256] fp32

    float* out  = (float*)d_out;               // [16,256,64,64] then [16] loss
    float* loss = out + (size_t)NB * CDIM * HW;

    __bf16* ewbf = (__bf16*)d_ws;                                   // 512 KB
    float*  enorm = (float*)((char*)d_ws + (size_t)NCODE * CDIM * sizeof(__bf16));

    vq_zero_loss<<<1, 32, 0, stream>>>(loss);
    vq_prep<<<NCODE, CDIM, 0, stream>>>(ew, ewbf, enorm);
    vq_main<<<NB * HDIM, 256, 0, stream>>>(ze, ew, ewbf, enorm, out, loss);
}
